// CustomMultiheadAttention_88158498718125
// MI455X (gfx1250) — matrix operand order repaired and checked by reading, not compiled, not run on hardware
//
#include <hip/hip_runtime.h>
#include <hip/hip_bf16.h>

#define BATCH 16
#define QKD 256
#define VD 512
#define QC 32
#define VC 64
#define BG 128      // BATCH*GROUPS
#define HW 4096     // 64*64

typedef __attribute__((ext_vector_type(16))) _Float16 v16h;
typedef __attribute__((ext_vector_type(8)))  _Float16 v8h;
typedef __attribute__((ext_vector_type(8)))  float    v8f;

__device__ __forceinline__ float sigmoidf_(float x) { return 1.f / (1.f + __expf(-x)); }

// ---------------------------------------------------------------- K0: out_w -> fp16
__global__ void k_convert_w(const float* __restrict__ wsrc, _Float16* __restrict__ wdst, int n)
{
    int i = blockIdx.x * blockDim.x + threadIdx.x;
    if (i < n) wdst[i] = (_Float16)wsrc[i];
}

// ---------------------------------------------------------------- K1: pooling + 1x1 conv + diag (per group)
__global__ void k_pool_conv1_diag(const float* __restrict__ q, const float* __restrict__ kk,
                                  const float* __restrict__ c1w, const float* __restrict__ c1b,
                                  const float* __restrict__ dw,  const float* __restrict__ db,
                                  float* __restrict__ hWp, float* __restrict__ wWp,
                                  float* __restrict__ sd)
{
    __shared__ float feat[QC * 128];  // [c][x]: x<64 row-mean over w, x>=64 col-mean over h
    __shared__ float df[QC];
    const int g = blockIdx.x;
    const int tid = threadIdx.x;
    const int b = g >> 3, gr = g & 7;
    const float* qbase = q + ((size_t)b * QKD + gr * QC) * HW;

    for (int task = tid; task < QC * 64; task += blockDim.x) {
        int c = task >> 6, h = task & 63;
        const float* row = qbase + (size_t)c * HW + h * 64;
        float s = 0.f;
        #pragma unroll 8
        for (int w = 0; w < 64; ++w) s += row[w];
        feat[c * 128 + h] = s * (1.f / 64.f);
    }
    for (int task = tid; task < QC * 64; task += blockDim.x) {
        int c = task >> 6, w = task & 63;
        const float* col = qbase + (size_t)c * HW + w;
        float s = 0.f;
        #pragma unroll 8
        for (int h = 0; h < 64; ++h) s += col[h * 64];
        feat[c * 128 + 64 + w] = s * (1.f / 64.f);
    }
    if (tid < QC) {
        const float* kc = kk + ((size_t)b * QKD + gr * QC + tid) * HW;
        float s = 0.f;
        for (int i = 0; i < 64; ++i) s += kc[i * 64 + i] + kc[i * 64 + (63 - i)];
        df[tid] = s * (1.f / 64.f);
    }
    __syncthreads();
    for (int task = tid; task < VC * 128; task += blockDim.x) {
        int o = task >> 7, x = task & 127;
        float s = c1b[o];
        #pragma unroll
        for (int c = 0; c < QC; ++c) s += c1w[o * QC + c] * feat[c * 128 + x];
        float sg = sigmoidf_(s);
        if (x < 64) hWp[((size_t)g * VC + o) * 64 + x] = sg;
        else        wWp[((size_t)g * VC + o) * 64 + (x - 64)] = sg;
    }
    if (tid < VC) {
        float s = db[tid];
        #pragma unroll
        for (int c = 0; c < QC; ++c) s += dw[tid * QC + c] * df[c];
        sd[(size_t)g * VC + tid] = sigmoidf_(s);
    }
}

// ---------------------------------------------------------------- K2: x1 channel sums / sumsq (x1 never stored)
__global__ void k_x1_stats(const float* __restrict__ v, const float* __restrict__ hWp,
                           const float* __restrict__ wWp, const float* __restrict__ sd,
                           float* __restrict__ S1, float* __restrict__ S2)
{
    __shared__ float hws[64], wws[64];
    __shared__ float red1[256], red2[256];
    const int gc = blockIdx.x;          // g*64 + c
    const int g = gc >> 6, c = gc & 63;
    const int tid = threadIdx.x;
    if (tid < 64) { hws[tid] = hWp[(size_t)gc * 64 + tid]; wws[tid] = wWp[(size_t)gc * 64 + tid]; }
    __syncthreads();
    const float sdc = sd[gc];
    const float* vc_ = v + ((size_t)(g >> 3) * VD + (g & 7) * VC + c) * HW;
    float s1 = 0.f, s2 = 0.f;
    for (int p = tid; p < HW; p += 256) {
        float x = vc_[p] * (hws[p >> 6] * wws[p & 63] + sdc);
        s1 += x; s2 += x * x;
    }
    red1[tid] = s1; red2[tid] = s2;
    __syncthreads();
    for (int off = 128; off > 0; off >>= 1) {
        if (tid < off) { red1[tid] += red1[tid + off]; red2[tid] += red2[tid + off]; }
        __syncthreads();
    }
    if (tid == 0) { S1[gc] = red1[0]; S2[gc] = red2[0]; }
}

// ---------------------------------------------------------------- K3: GN stats + softmax a1 + folded conv weights
__global__ void k_softmax_weff(const float* __restrict__ S1, const float* __restrict__ S2,
                               const float* __restrict__ gnw, const float* __restrict__ gnb,
                               const float* __restrict__ c3w, const float* __restrict__ c3b,
                               float* __restrict__ weff, float* __restrict__ beff)
{
    __shared__ float a1s[VC];
    __shared__ float ms[VC];
    __shared__ float stat[2];
    const int g = blockIdx.x, tid = threadIdx.x;
    if (tid == 0) {
        float t1 = 0.f, t2 = 0.f;
        for (int c = 0; c < VC; ++c) { t1 += S1[g * VC + c]; t2 += S2[g * VC + c]; }
        float mu  = t1 / (float)(VC * HW);
        float var = t2 / (float)(VC * HW) - mu * mu;
        stat[0] = mu; stat[1] = rsqrtf(var + 1e-5f);
    }
    __syncthreads();
    if (tid < VC)
        ms[tid] = (S1[g * VC + tid] * (1.f / (float)HW) - stat[0]) * stat[1] * gnw[tid] + gnb[tid];
    __syncthreads();
    if (tid == 0) {
        float mx = ms[0];
        for (int c = 1; c < VC; ++c) mx = fmaxf(mx, ms[c]);
        float se = 0.f;
        for (int c = 0; c < VC; ++c) { float e = __expf(ms[c] - mx); a1s[c] = e; se += e; }
        float inv = 1.f / se;
        float bb = 0.f;
        for (int c = 0; c < VC; ++c) { a1s[c] *= inv; bb += a1s[c] * c3b[c]; }
        beff[g] = bb;
    }
    __syncthreads();
    for (int t = tid; t < VC * 9; t += blockDim.x) {
        int i = t / 9, tap = t % 9;
        float s = 0.f;
        #pragma unroll
        for (int o = 0; o < VC; ++o) s += a1s[o] * c3w[(o * VC + i) * 9 + tap];
        weff[(size_t)g * VC * 9 + t] = s;
    }
}

// ---------------------------------------------------------------- K4: folded 3x3 conv -> sigmoid gate -> fp16 staging
// Writes Xh POSITION-MAJOR: Xh[((b*4096)+p)*512 + channel]  (so GEMM A-tiles are contiguous in K)
__global__ void k_att_gate(const float* __restrict__ v, const float* __restrict__ weff,
                           const float* __restrict__ beff, _Float16* __restrict__ Xh)
{
    __shared__ float we[VC * 9];
    __shared__ float att[256];
    const int bid = blockIdx.x;
    const int g = bid >> 4, h0 = (bid & 15) * 4;   // 4-row strip
    const int tid = threadIdx.x;
    const int b = g >> 3, gr = g & 7;
    const float* vg = v + ((size_t)b * VD + gr * VC) * HW;
    for (int t = tid; t < VC * 9; t += 256) we[t] = weff[(size_t)g * VC * 9 + t];
    __syncthreads();
    const int hloc = tid >> 6, w = tid & 63;
    const int h = h0 + hloc;
    float s = beff[g];
    for (int c = 0; c < VC; ++c) {
        const float* vc_ = vg + (size_t)c * HW;
        const float* wr = &we[c * 9];
        #pragma unroll
        for (int dy = 0; dy < 3; ++dy) {
            int hh = h + dy - 1;
            if ((unsigned)hh >= 64u) continue;
            const float* row = vc_ + hh * 64;
            #pragma unroll
            for (int dx = 0; dx < 3; ++dx) {
                int wp = w + dx - 1;
                if ((unsigned)wp < 64u) s += wr[dy * 3 + dx] * row[wp];
            }
        }
    }
    att[tid] = sigmoidf_(s);
    __syncthreads();
    const float gate = att[tid];
    const int p = h0 * 64 + tid;                          // position within image
    _Float16* dst = Xh + ((size_t)b * HW + p) * VD + gr * VC;  // 64 contiguous channels
    #pragma unroll
    for (int u = 0; u < 8; ++u) {
        v8h pack;
        #pragma unroll
        for (int j = 0; j < 8; ++j)
            pack[j] = (_Float16)(vg[(size_t)(u * 8 + j) * HW + p] * gate);
        *(v8h*)(dst + u * 8) = pack;
    }
}

// ---------------------------------------------------------------- K5: channel mix as D[p,o] = X^T * W^T via WMMA
// A = Xh [p][c] (M=positions), B = Wh [o][c] read as K-major (N=o). All fragments are
// two contiguous b128 loads per lane; no LDS, no barriers in the K-loop.
__global__ void __launch_bounds__(256) k_gemm_out(const _Float16* __restrict__ Wh,
                                                  const _Float16* __restrict__ Xh,
                                                  const float* __restrict__ ob,
                                                  float* __restrict__ out)
{
    const int bid = blockIdx.x;
    const int nt = bid & 7;           // o-tile of 64
    const int mt = (bid >> 3) & 31;   // position-tile of 128
    const int bi = bid >> 8;          // image
    const int tid = threadIdx.x, lane = tid & 31, wave = tid >> 5;
    const int m0 = mt * 128 + wave * 16;   // this wave's 16 positions
    const int n0 = nt * 64;                // shared o range

    v8f acc[4] = { {}, {}, {}, {} };
    const int ml = lane & 15, sel = lane >> 4, nl = lane & 15;
    // A: row p = m0+ml, elements K = k0 + sel*8 + j  /  k0 + 16 + sel*8 + j
    const _Float16* aptr = Xh + ((size_t)bi * HW + m0 + ml) * VD + sel * 8;
    // B: row o = n0 + t*16 + nl, elements K = k0 + sel*8 + j  /  k0 + 16 + sel*8 + j
    const _Float16* bbase = Wh + (size_t)n0 * VD + sel * 8;

    for (int k0 = 0; k0 < VD; k0 += 32) {
        v16h a;
        *((uint4*)&a)     = *(const uint4*)(aptr + k0);
        *((uint4*)&a + 1) = *(const uint4*)(aptr + k0 + 16);
        #pragma unroll
        for (int t = 0; t < 4; ++t) {
            v16h bf;
            const _Float16* bp = bbase + (size_t)(t * 16 + nl) * VD + k0;
            *((uint4*)&bf)     = *(const uint4*)bp;
            *((uint4*)&bf + 1) = *(const uint4*)(bp + 16);
            acc[t] = __builtin_amdgcn_wmma_f32_16x16x32_f16(
                false, a, false, bf, (short)0, acc[t], false, false);
        }
    }

    // D layout: m = r + 8*(lane>>4) (position), n = lane&15 (o). Each lane: 8 consecutive
    // positions at fixed o -> two float4 stores; bias is scalar per lane.
    #pragma unroll
    for (int t = 0; t < 4; ++t) {
        const int o = n0 + t * 16 + nl;
        const float bias = ob[o];
        float* dst = out + ((size_t)bi * VD + o) * HW + m0 + sel * 8;
        float4 lo = { acc[t][0] + bias, acc[t][1] + bias, acc[t][2] + bias, acc[t][3] + bias };
        float4 hi = { acc[t][4] + bias, acc[t][5] + bias, acc[t][6] + bias, acc[t][7] + bias };
        *(float4*)(dst)     = lo;
        *(float4*)(dst + 4) = hi;
    }
}

// ----------------------------------------------------------------
extern "C" void kernel_launch(void* const* d_in, const int* in_sizes, int n_in,
                              void* d_out, int out_size, void* d_ws, size_t ws_size,
                              hipStream_t stream)
{
    const float* q   = (const float*)d_in[0];
    const float* k   = (const float*)d_in[1];
    const float* v   = (const float*)d_in[2];
    const float* c1w = (const float*)d_in[3];
    const float* c1b = (const float*)d_in[4];
    const float* c3w = (const float*)d_in[5];
    const float* c3b = (const float*)d_in[6];
    const float* gnw = (const float*)d_in[7];
    const float* gnb = (const float*)d_in[8];
    const float* dw  = (const float*)d_in[9];
    const float* db  = (const float*)d_in[10];
    const float* ow  = (const float*)d_in[11];
    const float* ob  = (const float*)d_in[12];
    float* out = (float*)d_out;

    char* ws = (char*)d_ws;
    float*    hWp  = (float*)(ws + 0);                          // 2 MB
    float*    wWp  = (float*)(ws + (2u << 20));                 // 2 MB
    float*    sd   = (float*)(ws + (4u << 20));                 // 32 KB
    float*    S1   = (float*)(ws + (4u << 20) + 32768);         // 32 KB
    float*    S2   = (float*)(ws + (4u << 20) + 65536);         // 32 KB
    float*    weff = (float*)(ws + (4u << 20) + 98304);         // 288 KB
    float*    beff = (float*)(ws + (4u << 20) + 98304 + 294912);// 512 B
    _Float16* Wh   = (_Float16*)(ws + (5u << 20));              // 512 KB
    _Float16* Xh   = (_Float16*)(ws + (8u << 20));              // 64 MB

    hipLaunchKernelGGL(k_convert_w, dim3(1024), dim3(256), 0, stream, ow, Wh, 512 * 512);
    hipLaunchKernelGGL(k_pool_conv1_diag, dim3(BG), dim3(256), 0, stream,
                       q, k, c1w, c1b, dw, db, hWp, wWp, sd);
    hipLaunchKernelGGL(k_x1_stats, dim3(BG * VC), dim3(256), 0, stream,
                       v, hWp, wWp, sd, S1, S2);
    hipLaunchKernelGGL(k_softmax_weff, dim3(BG), dim3(256), 0, stream,
                       S1, S2, gnw, gnb, c3w, c3b, weff, beff);
    hipLaunchKernelGGL(k_att_gate, dim3(BG * 16), dim3(256), 0, stream,
                       v, weff, beff, Xh);
    hipLaunchKernelGGL(k_gemm_out, dim3(BATCH * 8 * 32), dim3(256), 0, stream,
                       Wh, Xh, ob, out);
}
